// CaDDN_Core_352187318845
// MI455X (gfx1250) — matrix operand order repaired and checked by reading, not compiled, not run on hardware
//
#include <hip/hip_runtime.h>
#include <hip/hip_bf16.h>
#include <math.h>

typedef __attribute__((ext_vector_type(16))) _Float16 v16h;
typedef __attribute__((ext_vector_type(8)))  float    v8f;

#define IMG_H 96
#define IMG_W 320
#define NPIX  (IMG_H * IMG_W)      // 30720
#define GX 200
#define GY 200
#define GZ 16
#define NBEV (GX * GY)             // 40000
#define DB 80
#define FC 64
#define CBEV 1024                  // FC*GZ
#define MOUT 128
#define KCHUNKS 288                // 9 taps * 32 channel-groups

// Zero-padded voxel tensor: [32 cg][YPAD][XPAD][32 c32], halo of 1 in y/x plus
// x-overhang slack for the last 32-pixel strip (x up to 223 -> xp up to 225).
#define YPAD 202
#define XPAD 232
#define VOXT_HALVES ((size_t)32 * YPAD * XPAD * 32)   // 47,988,736 halves = 96 MB

// ---------------------------------------------------------------------------
// Generic 3-input-channel 3x3 SAME conv (conv1 and depth head)
// ---------------------------------------------------------------------------
__global__ void k_conv3(const float* __restrict__ img, const float* __restrict__ w,
                        const float* __restrict__ bias, float* __restrict__ out, int CO) {
    int idx = blockIdx.x * blockDim.x + threadIdx.x;
    int tot = CO * NPIX;
    if (idx >= tot) return;
    int wp = idx % IMG_W;
    int hp = (idx / IMG_W) % IMG_H;
    int co = idx / NPIX;
    float acc = bias[co];
    #pragma unroll
    for (int ci = 0; ci < 3; ++ci) {
        #pragma unroll
        for (int ky = 0; ky < 3; ++ky) {
            int h = hp + ky - 1;
            if (h < 0 || h >= IMG_H) continue;
            #pragma unroll
            for (int kx = 0; kx < 3; ++kx) {
                int ww = wp + kx - 1;
                if (ww < 0 || ww >= IMG_W) continue;
                acc += w[((co * 3 + ci) * 3 + ky) * 3 + kx] * img[(ci * IMG_H + h) * IMG_W + ww];
            }
        }
    }
    out[idx] = acc;
}

// ---------------------------------------------------------------------------
// Per-channel mean/var (biased): one block per channel
// ---------------------------------------------------------------------------
__global__ void k_stats(const float* __restrict__ x, float* __restrict__ stats, int N) {
    int c = blockIdx.x;
    const float* p = x + (size_t)c * N;
    float s = 0.f, s2 = 0.f;
    for (int i = threadIdx.x; i < N; i += blockDim.x) {
        float v = p[i];
        s += v; s2 += v * v;
    }
    __shared__ float sh[256], sh2[256];
    sh[threadIdx.x] = s; sh2[threadIdx.x] = s2;
    __syncthreads();
    for (int off = 128; off > 0; off >>= 1) {
        if ((int)threadIdx.x < off) {
            sh[threadIdx.x]  += sh[threadIdx.x + off];
            sh2[threadIdx.x] += sh2[threadIdx.x + off];
        }
        __syncthreads();
    }
    if (threadIdx.x == 0) {
        float mean = sh[0] / (float)N;
        float var  = sh2[0] / (float)N - mean * mean;
        stats[c * 2]     = mean;
        stats[c * 2 + 1] = var;
    }
}

// ---------------------------------------------------------------------------
// y = relu(gamma*(x-mean)*rsqrt(var+eps)+beta)
// ---------------------------------------------------------------------------
__global__ void k_bnrelu(const float* __restrict__ x, const float* __restrict__ stats,
                         const float* __restrict__ g, const float* __restrict__ b,
                         float* __restrict__ y, int N, int C) {
    size_t i = (size_t)blockIdx.x * blockDim.x + threadIdx.x;
    size_t tot = (size_t)C * N;
    if (i >= tot) return;
    int c = (int)(i / N);
    float mean = stats[c * 2], var = stats[c * 2 + 1];
    float v = (x[i] - mean) * rsqrtf(var + 1e-5f);
    v = g[c] * v + b[c];
    y[i] = v > 0.f ? v : 0.f;
}

// ---------------------------------------------------------------------------
// Channel softmax over 80 depth bins, per pixel
// ---------------------------------------------------------------------------
__global__ void k_softmax80(const float* __restrict__ logits, float* __restrict__ probs) {
    int p = blockIdx.x * blockDim.x + threadIdx.x;
    if (p >= NPIX) return;
    float mx = -1e30f;
    for (int d = 0; d < DB; ++d) mx = fmaxf(mx, logits[d * NPIX + p]);
    float s = 0.f;
    for (int d = 0; d < DB; ++d) s += __expf(logits[d * NPIX + p] - mx);
    float inv = 1.0f / s;
    for (int d = 0; d < DB; ++d) probs[d * NPIX + p] = __expf(logits[d * NPIX + p] - mx) * inv;
}

// ---------------------------------------------------------------------------
// Zero-fill (16B per thread) for the padded voxel tensor
// ---------------------------------------------------------------------------
__global__ void k_fill0(uint4* __restrict__ p, size_t n16) {
    size_t i = (size_t)blockIdx.x * blockDim.x + threadIdx.x;
    if (i < n16) p[i] = make_uint4(0u, 0u, 0u, 0u);
}

// ---------------------------------------------------------------------------
// Voxel sampler (frustum never materialized): writes interior of padded
// channel-tiled f16 tensor voxt[cg][y+1][x+1][c32], channel = c*16 + z.
// ---------------------------------------------------------------------------
__global__ void k_sample(const float* __restrict__ feats, const float* __restrict__ dprobs,
                         const float* __restrict__ calib, _Float16* __restrict__ voxt) {
    int n = blockIdx.x * blockDim.x + threadIdx.x;
    if (n >= GZ * NBEV) return;
    int x = n % GX;
    int y = (n / GX) % GY;
    int z = n / NBEV;

    float X = x * (70.4f / 199.f);
    float Y = -40.f + y * (80.f / 199.f);
    float Z = -3.f + z * (4.f / 15.f);

    float u_ = calib[0] * X + calib[1] * Y + calib[2]  * Z + calib[3];
    float v_ = calib[4] * X + calib[5] * Y + calib[6]  * Z + calib[7];
    float d_ = calib[8] * X + calib[9] * Y + calib[10] * Z + calib[11];
    float invd = 1.0f / (d_ + 1e-5f);
    float u = u_ * invd, v = v_ * invd;
    float un = u / 319.f * 2.f - 1.f;
    float vn = v / 95.f  * 2.f - 1.f;
    float dn = (d_ - 2.f) / 40.f * 2.f - 1.f;
    float gx = ((un + 1.f) * 320.f - 1.f) * 0.5f;
    float gy = ((vn + 1.f) * 96.f  - 1.f) * 0.5f;
    float gz = ((dn + 1.f) * 80.f  - 1.f) * 0.5f;

    float x0f = floorf(gx), y0f = floorf(gy), z0f = floorf(gz);
    float wx = gx - x0f, wy = gy - y0f, wz = gz - z0f;
    int x0 = (int)x0f, y0 = (int)y0f, z0 = (int)z0f;

    float coef[4];
    int   off[4];
    int c = 0;
    #pragma unroll
    for (int dy = 0; dy < 2; ++dy) {
        #pragma unroll
        for (int dx = 0; dx < 2; ++dx) {
            int iy = y0 + dy, ix = x0 + dx;
            float fy = dy ? wy : 1.f - wy;
            float fx = dx ? wx : 1.f - wx;
            bool okhw = (iy >= 0 && iy < IMG_H && ix >= 0 && ix < IMG_W);
            int iyc = min(max(iy, 0), IMG_H - 1);
            int ixc = min(max(ix, 0), IMG_W - 1);
            float dpsum = 0.f;
            if (okhw) {
                if (z0 >= 0 && z0 < DB)         dpsum += (1.f - wz) * dprobs[(z0 * IMG_H + iyc) * IMG_W + ixc];
                if (z0 + 1 >= 0 && z0 + 1 < DB) dpsum += wz        * dprobs[((z0 + 1) * IMG_H + iyc) * IMG_W + ixc];
            }
            coef[c] = fy * fx * dpsum;
            off[c]  = iyc * IMG_W + ixc;
            ++c;
        }
    }

    for (int ch = 0; ch < FC; ++ch) {
        const float* f = feats + (size_t)ch * NPIX;
        float val = coef[0] * f[off[0]] + coef[1] * f[off[1]] +
                    coef[2] * f[off[2]] + coef[3] * f[off[3]];
        int cc = ch * GZ + z;                          // BEV channel = c*16 + z
        voxt[(((size_t)(cc >> 5) * YPAD + (y + 1)) * XPAD + (x + 1)) * 32 + (cc & 31)]
            = (_Float16)val;
    }
}

// ---------------------------------------------------------------------------
// Pack bev_w (fp32 [128][1024][3][3]) into WMMA A-fragment order, f16.
// wpack[mt][kc][lane][16]; kc = tap*32 + cg; A 16x32 in contiguous K order (same as B):
//   row M = mt*16 + (lane&15);  half j -> K = j + (lane>=16?16:0)
// ---------------------------------------------------------------------------
__global__ void k_packw(const float* __restrict__ bw, _Float16* __restrict__ wpack) {
    int t = blockIdx.x * blockDim.x + threadIdx.x;
    if (t >= 8 * KCHUNKS * 32 * 16) return;
    int j    = t & 15;
    int lane = (t >> 4) & 31;
    int kc   = (t >> 9) % KCHUNKS;
    int mt   = (t >> 9) / KCHUNKS;
    int K  = j + (lane >= 16 ? 16 : 0);
    int cg = kc & 31;
    int tap = kc >> 5;
    int ky = tap / 3, kx = tap % 3;
    int ci = cg * 32 + K;
    int m  = mt * 16 + (lane & 15);
    wpack[t] = (_Float16)bw[((m * CBEV + ci) * 3 + ky) * 3 + kx];
}

// ---------------------------------------------------------------------------
// BEV 3x3 conv as implicit GEMM, f16 WMMA + f32 accumulate.
// 2x2 register blocking: each wave owns 2 m-tiles x 2 n-tiles (4 v8f accs),
// so each k-chunk iteration = 2 A loads + 2 B loads -> 4 WMMAs (2 b128/WMMA).
// WG = 4 waves (128 thr) covering all 128 output channels on a 32-pixel strip.
// All B loads unconditional thanks to the zero-padded voxel tensor.
// Grid: 200 rows * 7 strips = 1400 WGs.
// ---------------------------------------------------------------------------
__global__ __launch_bounds__(128) void k_bev_gemm(
        const _Float16* __restrict__ voxt,   // [32][YPAD][XPAD][32], zero-padded
        const _Float16* __restrict__ wpack,  // [8][288][32][16]
        const float* __restrict__ bias,      // [128]
        float* __restrict__ bevraw) {        // [128][200][200]
    int wg = blockIdx.x;
    int y  = wg / 7;
    int x0 = (wg % 7) * 32;
    int wave = threadIdx.x >> 5;   // 0..3 -> m-tile pair {2*wave, 2*wave+1}
    int lane = threadIdx.x & 31;
    int n    = lane & 15;
    int klo  = lane & 16;          // 0 | 16 : K-half of the B fragment

    v8f acc00 = {}, acc01 = {}, acc10 = {}, acc11 = {};
    const size_t MT_STRIDE = (size_t)KCHUNKS * 512;       // halves per m-tile
    const _Float16* aptr = wpack + (size_t)(wave * 2) * MT_STRIDE + lane * 16;
    const size_t cg_stride = (size_t)YPAD * XPAD * 32;

    for (int tap = 0; tap < 9; ++tap) {
        int ky = tap / 3, kx = tap % 3;
        int yp = y + ky;            // padded y index of (y + ky - 1)
        int xp = x0 + n + kx;       // padded x index of (x0 + n + kx - 1)
        const _Float16* bptr = voxt + (((size_t)yp * XPAD + xp) * 32 + klo);
        #pragma unroll 2
        for (int cg = 0; cg < 32; ++cg) {
            v16h a0 = *(const v16h*)aptr;
            v16h a1 = *(const v16h*)(aptr + MT_STRIDE);   // +294912 B const offset
            aptr += 512;
            v16h b0 = *(const v16h*)bptr;
            v16h b1 = *(const v16h*)(bptr + 16 * 32);     // +1024 B const offset
            __builtin_prefetch(bptr + 2 * cg_stride, 0, 1);
            bptr += cg_stride;
            acc00 = __builtin_amdgcn_wmma_f32_16x16x32_f16(
                        false, a0, false, b0, (short)0, acc00, false, false);
            acc01 = __builtin_amdgcn_wmma_f32_16x16x32_f16(
                        false, a0, false, b1, (short)0, acc01, false, false);
            acc10 = __builtin_amdgcn_wmma_f32_16x16x32_f16(
                        false, a1, false, b0, (short)0, acc10, false, false);
            acc11 = __builtin_amdgcn_wmma_f32_16x16x32_f16(
                        false, a1, false, b1, (short)0, acc11, false, false);
        }
    }

    int mA = (wave * 2) * 16 + ((lane < 16) ? 0 : 8);   // m-tile 2*wave
    int mB = mA + 16;                                   // m-tile 2*wave+1
    int xa = x0 + n;
    int xb = xa + 16;
    #pragma unroll
    for (int r = 0; r < 8; ++r) {
        int ma = mA + r, mb = mB + r;
        float ba = bias[ma], bb = bias[mb];
        if (xa < GX) {
            bevraw[(size_t)ma * NBEV + y * GX + xa] = acc00[r] + ba;
            bevraw[(size_t)mb * NBEV + y * GX + xa] = acc10[r] + bb;
        }
        if (xb < GX) {
            bevraw[(size_t)ma * NBEV + y * GX + xb] = acc01[r] + ba;
            bevraw[(size_t)mb * NBEV + y * GX + xb] = acc11[r] + bb;
        }
    }
}

// ---------------------------------------------------------------------------
// Host-side orchestration
// ---------------------------------------------------------------------------
extern "C" void kernel_launch(void* const* d_in, const int* in_sizes, int n_in,
                              void* d_out, int out_size, void* d_ws, size_t ws_size,
                              hipStream_t stream) {
    const float* images  = (const float*)d_in[0];
    const float* calib   = (const float*)d_in[1];
    const float* conv1_w = (const float*)d_in[2];
    const float* conv1_b = (const float*)d_in[3];
    const float* bn1_g   = (const float*)d_in[4];
    const float* bn1_b   = (const float*)d_in[5];
    const float* depth_w = (const float*)d_in[6];
    const float* depth_b = (const float*)d_in[7];
    const float* bev_w   = (const float*)d_in[8];
    const float* bev_b   = (const float*)d_in[9];
    const float* bn2_g   = (const float*)d_in[10];
    const float* bn2_b   = (const float*)d_in[11];

    float* out  = (float*)d_out;
    float* bev  = out;                       // [128][200][200]
    float* dlog = out + (size_t)MOUT * NBEV; // [80][96][320]

    char* ws = (char*)d_ws;
    size_t o = 0;
    float*    conv1raw = (float*)(ws + o);    o += (size_t)FC * NPIX * 4;         // 7.86 MB
    float*    feats    = (float*)(ws + o);    o += (size_t)FC * NPIX * 4;         // 7.86 MB
    float*    dprobs   = (float*)(ws + o);    o += (size_t)DB * NPIX * 4;         // 9.83 MB
    _Float16* voxt     = (_Float16*)(ws + o); o += VOXT_HALVES * 2;               // 96.0 MB
    _Float16* wpack    = (_Float16*)(ws + o); o += (size_t)8 * KCHUNKS * 512 * 2; // 2.36 MB
    float*    bevraw   = (float*)(ws + o);    o += (size_t)MOUT * NBEV * 4;       // 20.5 MB
    float*    stats1   = (float*)(ws + o);    o += 64 * 2 * 4;
    float*    stats2   = (float*)(ws + o);    o += 128 * 2 * 4;
    (void)ws_size; (void)in_sizes; (void)n_in; (void)out_size;

    // 0) zero the padded voxel tensor (halo must be exactly 0)
    size_t n16 = VOXT_HALVES / 8;
    k_fill0<<<(unsigned)((n16 + 255) / 256), 256, 0, stream>>>((uint4*)voxt, n16);
    // 1) conv1 raw
    k_conv3<<<(FC * NPIX + 255) / 256, 256, 0, stream>>>(images, conv1_w, conv1_b, conv1raw, FC);
    // 2) BN1 stats + apply
    k_stats<<<FC, 256, 0, stream>>>(conv1raw, stats1, NPIX);
    k_bnrelu<<<(FC * NPIX + 255) / 256, 256, 0, stream>>>(conv1raw, stats1, bn1_g, bn1_b, feats, NPIX, FC);
    // 3) depth logits (straight to output) + softmax probs
    k_conv3<<<(DB * NPIX + 255) / 256, 256, 0, stream>>>(images, depth_w, depth_b, dlog, DB);
    k_softmax80<<<(NPIX + 255) / 256, 256, 0, stream>>>(dlog, dprobs);
    // 4) lift to padded channel-tiled f16 voxel tensor
    k_sample<<<(GZ * NBEV + 255) / 256, 256, 0, stream>>>(feats, dprobs, calib, voxt);
    // 5) pack BEV weights into A-fragment layout
    k_packw<<<(8 * KCHUNKS * 32 * 16) / 256, 256, 0, stream>>>(bev_w, wpack);
    // 6) BEV conv as WMMA implicit GEMM (94.4 GFLOP — dominant cost)
    k_bev_gemm<<<GY * 7, 128, 0, stream>>>(voxt, wpack, bev_b, bevraw);
    // 7) BN2 stats + apply -> final BEV output
    k_stats<<<MOUT, 256, 0, stream>>>(bevraw, stats2, NBEV);
    k_bnrelu<<<(MOUT * NBEV + 255) / 256, 256, 0, stream>>>(bevraw, stats2, bn2_g, bn2_b, bev, NBEV, MOUT);
}
